// RPNModel_9552007266767
// MI455X (gfx1250) — matrix operand order repaired and checked by reading, not compiled, not run on hardware
//
#include <hip/hip_runtime.h>
#include <hip/hip_bf16.h>
#include <math.h>

// ---------------------------------------------------------------------------
// Types for CDNA5 WMMA (wave32): 16x16x32 bf16 -> f32
// ---------------------------------------------------------------------------
typedef __attribute__((ext_vector_type(16))) __bf16 v16bf;
typedef __attribute__((ext_vector_type(8)))  float  v8f;
typedef __attribute__((ext_vector_type(4)))  int    v4i;

typedef __attribute__((address_space(1))) v4i gv4i;   // global int4
typedef __attribute__((address_space(3))) v4i lv4i;   // LDS int4

union Frag32B {       // 32 bytes = 16 bf16 = 8 VGPRs
    v16bf v;
    uint4 q[2];
};

// Problem constants
#define BATCH   8
#define CIN     1024
#define HH      50
#define WW      50
#define HP      52                   // padded height (1-pixel zero halo)
#define WP      52                   // padded width
#define NPIX    (HH * WW)            // 2500
#define MTOT    (BATCH * NPIX)       // 20000
#define COUT    512
#define KTOT    (9 * CIN)            // 9216
#define NA      9
#define NPROP   (NPIX * NA)          // 22500
#define PRE_N   2000
#define POST_N  100
#define IMGW    1600.0f
#define IMGH    1600.0f
#define CLIPV   4.135166556742356f   // log(1000/16)

#define WAVES   8                    // waves per block in the GEMM
#define MROWS   32                   // M rows per wave (2 tiles of 16)
#define MGRPS   (MTOT / MROWS)       // 625 wave-groups

// ---------------------------------------------------------------------------
// K0: features NCHW f32 -> padded NHWC bf16
//     out[((b*52 + y+1)*52 + x+1)*1024 + c]; halo cells = 0
// ---------------------------------------------------------------------------
__global__ __launch_bounds__(256) void pack_feat(const float* __restrict__ in,
                                                 __bf16* __restrict__ out) {
    long tid = (long)blockIdx.x * blockDim.x + threadIdx.x;   // padded flat idx
    if (tid >= (long)BATCH * HP * WP * CIN) return;
    int c    = (int)(tid % CIN);
    long q   = tid / CIN;           // b*HP*WP + yy*WP + xx
    int xx   = (int)(q % WP);
    long q2  = q / WP;
    int yy   = (int)(q2 % HP);
    int b    = (int)(q2 / HP);
    const int y = yy - 1, x = xx - 1;
    __bf16 v = (__bf16)0.0f;
    if (y >= 0 && y < HH && x >= 0 && x < WW) {
        v = (__bf16)in[(((long)b * CIN + c) * NPIX) + y * WW + x];
    }
    out[tid] = v;
}

// ---------------------------------------------------------------------------
// K1: W1 (512,1024,3,3) f32 -> bf16 packed [n][p][c]  (9216 per n)
// ---------------------------------------------------------------------------
__global__ __launch_bounds__(256) void pack_w1(const float* __restrict__ in,
                                               __bf16* __restrict__ out) {
    long tid = (long)blockIdx.x * blockDim.x + threadIdx.x;   // output flat idx
    if (tid >= (long)COUT * KTOT) return;
    int n = (int)(tid / KTOT);
    int r = (int)(tid % KTOT);
    int p = r / CIN;
    int c = r % CIN;
    out[tid] = (__bf16)in[(((long)n * CIN + c) * 9) + p];
}

// ---------------------------------------------------------------------------
// Cooperative transfer of one 64n x 32k bf16 B slab (4 KB) into LDS.
// 256 threads, 16B each. Async global->LDS (ASYNCcnt) if builtin available.
// ---------------------------------------------------------------------------
#if __has_builtin(__builtin_amdgcn_global_load_async_to_lds_b128)
#define HAVE_ASYNC_LDS 1
#else
#define HAVE_ASYNC_LDS 0
#endif

__device__ __forceinline__ void stage_slab(const __bf16* __restrict__ gsrc,
                                           __bf16* __restrict__ ldst) {
#if HAVE_ASYNC_LDS
    __builtin_amdgcn_global_load_async_to_lds_b128((gv4i*)gsrc, (lv4i*)ldst, 0, 0);
#else
    const uint4 t = *(const uint4*)gsrc;
    *(uint4*)ldst = t;
#endif
}

__device__ __forceinline__ void stage_wait() {
#if HAVE_ASYNC_LDS
#if __has_builtin(__builtin_amdgcn_s_wait_asynccnt)
    __builtin_amdgcn_s_wait_asynccnt(0);
#else
    asm volatile("s_wait_asynccnt 0x0" ::: "memory");
#endif
#endif
}

// ---------------------------------------------------------------------------
// K2: implicit-GEMM 3x3 conv with v_wmma_f32_16x16x32_bf16 + bias + ReLU.
// Block = 256 threads = 8 waves sharing a 64-wide N group via an async-DMA
// double-buffered LDS B slab. Each wave owns TWO 16-row M tiles (8 acc),
// so the 4 B fragments are reused for 8 WMMAs per K-step. The zero-padded
// feature halo makes every A load unconditional (no EXEC divergence).
// grid = (ceil(625/8)=79, 8), block = 256
// ---------------------------------------------------------------------------
__global__ __launch_bounds__(256) void conv3x3_wmma(
    const __bf16* __restrict__ featB,   // [B][52][52][1024], zero halo
    const __bf16* __restrict__ wpack,   // [512][9][1024]
    const float*  __restrict__ bias1,   // [512]
    float*        __restrict__ T)       // [20000][512] fp32
{
    __shared__ uint4 shB4[2][256];      // 2 x 4 KB B slabs (64 n x 32 k bf16)

    const int tid  = threadIdx.x;
    const int lane = tid & 31;
    const int wave = tid >> 5;
    const int mrow = lane & 15;
    const int hi   = (lane >> 4) & 1;   // 0: lanes 0-15, 1: lanes 16-31

    int mg = blockIdx.x * WAVES + wave;
    if (mg >= MGRPS) mg = MGRPS - 1;    // benign duplicate work on the tail
    const int n0 = blockIdx.y * 64;

    // Two A rows per lane (M = lane&15 within each 16-row tile)
    const int m0 = mg * MROWS + mrow;
    const int m1 = m0 + 16;
    const int b0 = m0 / NPIX, pix0 = m0 % NPIX, ya = pix0 / WW, xa = pix0 % WW;
    const int b1 = m1 / NPIX, pix1 = m1 % NPIX, yb = pix1 / WW, xb = pix1 % WW;

    // A fragment K groups: lanes0-15 -> {K0..7, K16..23}; lanes16-31 -> {K8..15, K24..31}
    const int aoff = hi ? 8 : 0;
    // B fragment K groups: lanes0-15 -> {K0..7, K16..23}; lanes16-31 -> {K8..15, K24..31}
    const int koffB = hi ? 8 : 0;

    // Base of the 3x3 window in the padded image: tap p adds (p/3)*row + (p%3)
    const __bf16* awin0 =
        featB + (((size_t)b0 * HP + ya) * WP + xa) * CIN + aoff;   // (ya+dy+1-1, xa+dx+1-1)
    const __bf16* awin1 =
        featB + (((size_t)b1 * HP + yb) * WP + xb) * CIN + aoff;

    // Cooperative B loader: thread -> (n_local = tid>>2, k-part = (tid&3)*8)
    const int ln = tid >> 2;
    const int lk = (tid & 3) * 8;
    const __bf16* bsrc0 = wpack + (size_t)(n0 + ln) * KTOT + lk;  // + 32*s linear
    __bf16* const shBase = (__bf16*)shB4;
    __bf16* const sdst   = shBase + (size_t)(ln * 32 + lk);       // + buf*2048

    v8f acc0[4] = {v8f{}, v8f{}, v8f{}, v8f{}};
    v8f acc1[4] = {v8f{}, v8f{}, v8f{}, v8f{}};

    // -------- prologue: stage slab 0 --------
    stage_slab(bsrc0, sdst);
    stage_wait();
    __syncthreads();

    // -------- main K loop: 9 taps x 32 c0-steps --------
    for (int p = 0; p < 9; ++p) {
        // per-tap base pointers (halo padding -> always in bounds, zeros outside)
        const int dy = p / 3, dx = p % 3;           // 0..2 offsets in padded frame
        const __bf16* ab0 = awin0 + ((size_t)dy * WP + dx) * CIN;
        const __bf16* ab1 = awin1 + ((size_t)dy * WP + dx) * CIN;
        const __bf16* bstageP = bsrc0 + (size_t)p * CIN;   // DMA src, +32/step

        // A fragments for it = 0 of this tap (ping-pong slot 0)
        Frag32B a0[2], a1[2];
        a0[0].q[0] = *(const uint4*)(ab0);  a0[0].q[1] = *(const uint4*)(ab0 + 16);
        a1[0].q[0] = *(const uint4*)(ab1);  a1[0].q[1] = *(const uint4*)(ab1 + 16);

#pragma unroll 2
        for (int it = 0; it < 32; ++it) {
            const int cur = it & 1;          // LDS buffer & A slot (p*32 is even)
            const int nxt = cur ^ 1;

            // B fragments for this step from LDS (shared by both M tiles)
            const __bf16* sb = shBase + cur * 2048;
            Frag32B bf[4];
#pragma unroll
            for (int j = 0; j < 4; ++j) {
                const __bf16* bp = sb + (j * 16 + mrow) * 32 + koffB;
                bf[j].q[0] = *(const uint4*)(bp);
                bf[j].q[1] = *(const uint4*)(bp + 16);
            }

            // async-DMA the next 64n x 32k slab into the other LDS buffer
            {
                const __bf16* src = bstageP + (it + 1) * 32;
                if (p == 8 && it == 31) src -= 32;          // clamp; never consumed
                stage_slab(src, sdst + nxt * 2048);
            }

            // register-prefetch A fragments for it+1 (same tap, unconditional)
            if (it < 31) {
                const int c0n = (it + 1) * 32;
                a0[nxt].q[0] = *(const uint4*)(ab0 + c0n);
                a0[nxt].q[1] = *(const uint4*)(ab0 + c0n + 16);
                a1[nxt].q[0] = *(const uint4*)(ab1 + c0n);
                a1[nxt].q[1] = *(const uint4*)(ab1 + c0n + 16);
            }

            // 8 WMMAs on the matrix pipe (B fragments reused across M tiles)
#pragma unroll
            for (int j = 0; j < 4; ++j) {
                acc0[j] = __builtin_amdgcn_wmma_f32_16x16x32_bf16(
                    false, a0[cur].v, false, bf[j].v, (short)0, acc0[j], false, false);
            }
#pragma unroll
            for (int j = 0; j < 4; ++j) {
                acc1[j] = __builtin_amdgcn_wmma_f32_16x16x32_bf16(
                    false, a1[cur].v, false, bf[j].v, (short)0, acc1[j], false, false);
            }

            stage_wait();          // this wave's async slab portion landed
            __syncthreads();       // publish next slab / retire readers
        }
    }

    // Epilogue: C/D layout — lane0-15: (M=r, N=lane); lane16-31: (M=8+r, N=lane-16)
#pragma unroll
    for (int j = 0; j < 4; ++j) {
        const int n  = n0 + j * 16 + mrow;
        const float bv = bias1[n];
#pragma unroll
        for (int r = 0; r < 8; ++r) {
            const int moA = mg * MROWS + r + (hi ? 8 : 0);
            const int moB = moA + 16;
            float vA = acc0[j][r] + bv;
            float vB = acc1[j][r] + bv;
            T[(size_t)moA * COUT + n] = vA > 0.0f ? vA : 0.0f;
            T[(size_t)moB * COUT + n] = vB > 0.0f ? vB : 0.0f;
        }
    }
}

// ---------------------------------------------------------------------------
// K3: 1x1 heads + sigmoid + anchor decode + clip + validity mask.
// One thread per (b, pix, a) proposal.
// ---------------------------------------------------------------------------
__global__ __launch_bounds__(256) void heads_decode(
    const float* __restrict__ T,        // [20000][512]
    const float* __restrict__ W2, const float* __restrict__ b2,   // (36,512),(36)
    const float* __restrict__ W3, const float* __restrict__ b3,   // (9,512),(9)
    float* __restrict__ boxesBuf,       // [8][22500][4]
    float* __restrict__ scoresBuf)      // [8][22500] (masked)
{
    int tid = blockIdx.x * blockDim.x + threadIdx.x;
    if (tid >= MTOT * NA) return;
    const int m = tid / NA;
    const int a = tid % NA;
    const int b   = m / NPIX;
    const int pix = m % NPIX;
    const int y   = pix / WW;
    const int x   = pix % WW;

    const float* trow = T + (size_t)m * COUT;
    const float* w0 = W2 + (size_t)(a * 4 + 0) * COUT;
    const float* w1 = W2 + (size_t)(a * 4 + 1) * COUT;
    const float* w2 = W2 + (size_t)(a * 4 + 2) * COUT;
    const float* w3 = W2 + (size_t)(a * 4 + 3) * COUT;
    const float* wl = W3 + (size_t)a * COUT;
    float o0 = b2[a * 4 + 0], o1 = b2[a * 4 + 1];
    float o2 = b2[a * 4 + 2], o3 = b2[a * 4 + 3];
    float lg = b3[a];
    for (int c = 0; c < COUT; ++c) {
        const float tv = trow[c];
        o0 = fmaf(tv, w0[c], o0);
        o1 = fmaf(tv, w1[c], o1);
        o2 = fmaf(tv, w2[c], o2);
        o3 = fmaf(tv, w3[c], o3);
        lg = fmaf(tv, wl[c], lg);
    }

    // anchor (area index a/3, ratio index a%3)
    const float areas[3]  = {128.0f * 128.0f, 256.0f * 256.0f, 512.0f * 512.0f};
    const float ratios[3] = {0.5f, 1.0f, 2.0f};
    const float area = areas[a / 3], ratio = ratios[a % 3];
    const float wa = sqrtf(area * ratio);
    const float ha = sqrtf(area / ratio);
    const float cxa = ((float)x + 0.5f) * 32.0f;
    const float cya = ((float)y + 0.5f) * 32.0f;

    const float tw = fminf(o2, CLIPV);
    const float th = fminf(o3, CLIPV);
    const float px = fmaf(o0, wa, cxa);
    const float py = fmaf(o1, ha, cya);
    const float pw = expf(tw) * wa;
    const float ph = expf(th) * ha;

    float x0 = px - 0.5f * pw, y0 = py - 0.5f * ph;
    float x1 = px + 0.5f * pw, y1 = py + 0.5f * ph;
    x0 = fminf(fmaxf(x0, 0.0f), IMGW);  y0 = fminf(fmaxf(y0, 0.0f), IMGH);
    x1 = fminf(fmaxf(x1, 0.0f), IMGW);  y1 = fminf(fmaxf(y1, 0.0f), IMGH);
    const float bw = x1 - x0, bh = y1 - y0;

    const float sc = 1.0f / (1.0f + expf(-lg));
    const bool valid = (sc >= 0.1f) && (bw > 0.01f * IMGW) && (bh > 0.01f * IMGH);

    const int i = pix * NA + a;                    // reference index order
    const size_t o = (size_t)b * NPROP + i;
    boxesBuf[o * 4 + 0] = x0;
    boxesBuf[o * 4 + 1] = y0;
    boxesBuf[o * 4 + 2] = x1;
    boxesBuf[o * 4 + 3] = y1;
    scoresBuf[o] = valid ? sc : -1.0f;
}

// ---------------------------------------------------------------------------
// K4: per-batch top-2000 (iterative argmax, lowest-index tie-break like top_k)
// grid = 8 blocks, 256 threads. Mutates scoresBuf (rebuilt by K3 each call).
// ---------------------------------------------------------------------------
__global__ __launch_bounds__(256) void topk2000(
    float* __restrict__ scoresBuf,       // [8][22500]
    int*   __restrict__ topIdx,          // [8][2000]
    float* __restrict__ topS)            // [8][2000]
{
    __shared__ float sv[256];
    __shared__ int   si[256];
    const int b = blockIdx.x;
    const int tid = threadIdx.x;
    float* sc = scoresBuf + (size_t)b * NPROP;

    for (int t = 0; t < PRE_N; ++t) {
        float best = -1e30f;
        int   bi   = 0x7fffffff;
        for (int i = tid; i < NPROP; i += 256) {
            const float v = sc[i];
            if (v > best || (v == best && i < bi)) { best = v; bi = i; }
        }
        sv[tid] = best; si[tid] = bi;
        __syncthreads();
        for (int s = 128; s > 0; s >>= 1) {
            if (tid < s) {
                const float v2 = sv[tid + s];
                const int   i2 = si[tid + s];
                if (v2 > sv[tid] || (v2 == sv[tid] && i2 < si[tid])) {
                    sv[tid] = v2; si[tid] = i2;
                }
            }
            __syncthreads();
        }
        if (tid == 0) {
            const int idx = si[0];
            topIdx[b * PRE_N + t] = idx;
            topS[b * PRE_N + t]   = sv[0];
            sc[idx] = -3.0f;                 // exclude from further rounds
        }
        __syncthreads();
    }
}

// ---------------------------------------------------------------------------
// K5: per-batch NMS (sequential outer loop, parallel suppression) + top-100.
// grid = 8 blocks, 256 threads. LDS: 2000 boxes (32 KB) + keep flags (8 KB).
// ---------------------------------------------------------------------------
__global__ __launch_bounds__(256) void nms_top100(
    const float* __restrict__ boxesBuf,  // [8][22500][4]
    const int*   __restrict__ topIdx,    // [8][2000]
    const float* __restrict__ topS,      // [8][2000]
    float*       __restrict__ dout)      // boxes (8,100,4) then scores (8,100)
{
    __shared__ float bx[PRE_N][4];
    __shared__ int   keep[PRE_N];
    const int b = blockIdx.x;
    const int tid = threadIdx.x;

    for (int t = tid; t < PRE_N; t += 256) {
        const int idx = topIdx[b * PRE_N + t];
        const float* s = boxesBuf + ((size_t)b * NPROP + idx) * 4;
        bx[t][0] = s[0]; bx[t][1] = s[1]; bx[t][2] = s[2]; bx[t][3] = s[3];
        keep[t] = topS[b * PRE_N + t] > 0.0f ? 1 : 0;
    }
    __syncthreads();

    for (int i = 0; i < PRE_N; ++i) {
        if (keep[i]) {
            const float ax0 = bx[i][0], ay0 = bx[i][1];
            const float ax1 = bx[i][2], ay1 = bx[i][3];
            const float areaA = (ax1 - ax0) * (ay1 - ay0);
            for (int j = i + 1 + tid; j < PRE_N; j += 256) {
                if (keep[j]) {
                    const float bx0 = bx[j][0], by0 = bx[j][1];
                    const float bx1 = bx[j][2], by1 = bx[j][3];
                    const float areaB = (bx1 - bx0) * (by1 - by0);
                    const float ix0 = fmaxf(ax0, bx0), iy0 = fmaxf(ay0, by0);
                    const float ix1 = fminf(ax1, bx1), iy1 = fminf(ay1, by1);
                    const float iw = fmaxf(ix1 - ix0, 0.0f);
                    const float ih = fmaxf(iy1 - iy0, 0.0f);
                    const float inter = iw * ih;
                    const float iou = inter / (areaA + areaB - inter + 1e-9f);
                    if (iou > 0.7f) keep[j] = 0;
                }
            }
        }
        __syncthreads();
    }

    if (tid == 0) {
        float* ob = dout + (size_t)b * POST_N * 4;
        float* os = dout + (size_t)BATCH * POST_N * 4 + (size_t)b * POST_N;
        int cnt = 0;
        for (int t = 0; t < PRE_N && cnt < POST_N; ++t) {
            if (keep[t]) {
                ob[cnt * 4 + 0] = bx[t][0]; ob[cnt * 4 + 1] = bx[t][1];
                ob[cnt * 4 + 2] = bx[t][2]; ob[cnt * 4 + 3] = bx[t][3];
                os[cnt] = topS[b * PRE_N + t];
                ++cnt;
            }
        }
        for (int t = 0; t < PRE_N && cnt < POST_N; ++t) {
            if (!keep[t]) {
                ob[cnt * 4 + 0] = bx[t][0]; ob[cnt * 4 + 1] = bx[t][1];
                ob[cnt * 4 + 2] = bx[t][2]; ob[cnt * 4 + 3] = bx[t][3];
                os[cnt] = -1.0f;
                ++cnt;
            }
        }
    }
}

// ---------------------------------------------------------------------------
// Host: workspace layout + launches
// ---------------------------------------------------------------------------
extern "C" void kernel_launch(void* const* d_in, const int* in_sizes, int n_in,
                              void* d_out, int out_size, void* d_ws, size_t ws_size,
                              hipStream_t stream) {
    (void)in_sizes; (void)n_in; (void)out_size; (void)ws_size;
    const float* features = (const float*)d_in[0];
    const float* W1 = (const float*)d_in[1];
    const float* b1 = (const float*)d_in[2];
    const float* W2 = (const float*)d_in[3];
    const float* b2 = (const float*)d_in[4];
    const float* W3 = (const float*)d_in[5];
    const float* b3 = (const float*)d_in[6];
    float* dout = (float*)d_out;

    char* ws = (char*)d_ws;
    size_t off = 0;
    auto alloc = [&](size_t bytes) { char* p = ws + off; off = (off + bytes + 255) & ~(size_t)255; return p; };

    __bf16* featB   = (__bf16*)alloc((size_t)BATCH * HP * WP * CIN * sizeof(__bf16)); // 44.3 MB
    __bf16* wpack   = (__bf16*)alloc((size_t)COUT * KTOT * sizeof(__bf16));           //  9.44 MB
    float*  T       = (float*) alloc((size_t)MTOT * COUT * sizeof(float));            // 40.96 MB
    float*  boxesB  = (float*) alloc((size_t)BATCH * NPROP * 4 * sizeof(float));
    float*  scoresB = (float*) alloc((size_t)BATCH * NPROP * sizeof(float));
    int*    topIdx  = (int*)   alloc((size_t)BATCH * PRE_N * sizeof(int));
    float*  topS    = (float*) alloc((size_t)BATCH * PRE_N * sizeof(float));

    {   // K0: pack features into zero-haloed NHWC bf16
        const long n = (long)BATCH * HP * WP * CIN;
        pack_feat<<<dim3((unsigned)((n + 255) / 256)), dim3(256), 0, stream>>>(features, featB);
    }
    {   // K1: pack weights
        const long n = (long)COUT * KTOT;
        pack_w1<<<dim3((unsigned)((n + 255) / 256)), dim3(256), 0, stream>>>(W1, wpack);
    }
    // K2: WMMA conv — 8 waves/block, 2 M tiles/wave, async-DMA LDS B slab
    conv3x3_wmma<<<dim3((MGRPS + WAVES - 1) / WAVES, COUT / 64), dim3(256), 0, stream>>>(
        featB, wpack, b1, T);

    {   // K3: heads + decode
        const int n = MTOT * NA;
        heads_decode<<<dim3((n + 255) / 256), dim3(256), 0, stream>>>(T, W2, b2, W3, b3, boxesB, scoresB);
    }
    // K4: top-2000 per batch
    topk2000<<<dim3(BATCH), dim3(256), 0, stream>>>(scoresB, topIdx, topS);
    // K5: NMS + final top-100
    nms_top100<<<dim3(BATCH), dim3(256), 0, stream>>>(boxesB, topIdx, topS, dout);
}
